// SV_DGCNN_PSEG_mac_63007170232741
// MI455X (gfx1250) — matrix operand order repaired and checked by reading, not compiled, not run on hardware
//
#include <hip/hip_runtime.h>
#include <hip/hip_bf16.h>
#include <math.h>

typedef __attribute__((ext_vector_type(16))) _Float16 v16h;
typedef __attribute__((ext_vector_type(8)))  float    v8f;

// ============================================================================
// Generic WMMA GEMM:  Y[r,o] = act( sum_i X[r,i] * W[o,i] )
//   X = concat(X0: R x I0, X1: R x I1) row-major (X1 may be null with I1==0)
//   W row-major (O x (I0+I1)).  Optional batching via blockIdx.z + strides.
// Block = 256 threads = 8 waves; block tile 64(rows) x 64(cols); K-tile 64.
// Wave grid 4(M) x 2(N); each wave computes 16x32 = two 16x16 WMMA tiles
// sharing one A fragment; K-tile unrolled 2x -> 4 v_wmma per barrier pair.
// Staging is branch-free (clamped addresses + cndmask-zeroing) and the
// f16 LDS writes are packed into 16B ds_store_b128's.
// act: 0=none, 1=leaky_relu(0.2), 2=sigmoid
// ============================================================================
__global__ __launch_bounds__(256) void k_gemm(
    const float* __restrict__ X0, int I0,
    const float* __restrict__ X1, int I1,
    const float* __restrict__ W, float* __restrict__ Y,
    int R, int O, int act,
    long sX0, long sX1, long sW, long sY)
{
  const int bz = blockIdx.z;
  X0 += (long)bz * sX0;
  if (X1) X1 += (long)bz * sX1;
  W  += (long)bz * sW;
  Y  += (long)bz * sY;
  const int I = I0 + I1;

  // row pitch 72 halves = 144B (multiple of 16B) -> b128-aligned frag reads
  __shared__ __align__(16) _Float16 Xs[64][72];   // 64 rows x 64 k
  __shared__ __align__(16) _Float16 Ws[64][72];   // 64 cols x 64 k

  const int tid   = threadIdx.x;
  const int wave  = tid >> 5;
  const int lane  = tid & 31;
  const int waveM = wave & 3;       // 4 row tiles of 16
  const int waveN = wave >> 2;      // 2 col groups of 32
  const int rowBase = blockIdx.y * 64;
  const int colBase = blockIdx.x * 64;

  // staging: each thread owns one tile-row and 16 consecutive k
  const int rr = tid >> 2;          // 0..63
  const int kq = (tid & 3) * 16;    // 0,16,32,48

  const int rG = rowBase + rr;
  const int rC = (rG < R) ? rG : (R - 1);   // clamped -> always legal address
  const int oG = colBase + rr;
  const int oC = (oG < O) ? oG : (O - 1);

  v8f acc0 = {}, acc1 = {};

  for (int k0 = 0; k0 < I; k0 += 64) {
    // ---- prefetch next K tile into near caches (WGP scope) ----
    if (k0 + 64 < I) {
      int kp  = k0 + 64 + kq;
      int kpc = (kp < I) ? kp : (I - 1);
      if (kpc < I0) __builtin_prefetch(X0 + (long)rC * I0 + kpc, 0, 3);
      __builtin_prefetch(W + (long)oC * I + kpc, 0, 3);
    }

    // ---- stage X tile: unconditional loads, packed b128 LDS stores ----
    {
      float v[16];
#pragma unroll
      for (int u = 0; u < 16; ++u) {
        int k  = k0 + kq + u;
        int kc = (k < I) ? k : (I - 1);
        const float* p = (kc < I0) ? (X0 + (long)rC * I0 + kc)
                                   : (X1 + (long)rC * I1 + (kc - I0));
        v[u] = *p;
      }
      union { _Float16 h[8]; uint4 q; } pk;
#pragma unroll
      for (int u = 0; u < 8; ++u) {
        int k = k0 + kq + u;
        pk.h[u] = (_Float16)((rG < R && k < I) ? v[u] : 0.f);
      }
      *(uint4*)&Xs[rr][kq] = pk.q;
#pragma unroll
      for (int u = 0; u < 8; ++u) {
        int k = k0 + kq + 8 + u;
        pk.h[u] = (_Float16)((rG < R && k < I) ? v[8 + u] : 0.f);
      }
      *(uint4*)&Xs[rr][kq + 8] = pk.q;
    }
    // ---- stage W tile ----
    {
      float v[16];
#pragma unroll
      for (int u = 0; u < 16; ++u) {
        int k  = k0 + kq + u;
        int kc = (k < I) ? k : (I - 1);
        v[u] = W[(long)oC * I + kc];
      }
      union { _Float16 h[8]; uint4 q; } pk;
#pragma unroll
      for (int u = 0; u < 8; ++u) {
        int k = k0 + kq + u;
        pk.h[u] = (_Float16)((oG < O && k < I) ? v[u] : 0.f);
      }
      *(uint4*)&Ws[rr][kq] = pk.q;
#pragma unroll
      for (int u = 0; u < 8; ++u) {
        int k = k0 + kq + 8 + u;
        pk.h[u] = (_Float16)((oG < O && k < I) ? v[8 + u] : 0.f);
      }
      *(uint4*)&Ws[rr][kq + 8] = pk.q;
    }
    __syncthreads();

    // A fragment 16x32 f16 (this code's contiguous k order): lanes 0-15: M=lane, K 0-15;
    // lanes 16-31: M=lane-16, K 16-31.
    // B fragment 32x16 f16: lane = column; lanes 0-15 K 0-15, 16-31 K 16-31.
    const int mLoc  = waveM * 16 + (lane & 15);
    const int koffA = (lane >= 16) ? 16 : 0;
    const int nLoc  = waveN * 32 + (lane & 15);
    const int koffB = (lane >= 16) ? 16 : 0;
#pragma unroll
    for (int half = 0; half < 2; ++half) {
      const int kb = half * 32;
      v16h afrag, bf0, bf1;
#pragma unroll
      for (int h = 0; h < 16; ++h) {
        int k = kb + koffA + h;
        afrag[h] = Xs[mLoc][k];
      }
#pragma unroll
      for (int h = 0; h < 16; ++h) {
        bf0[h] = Ws[nLoc][kb + koffB + h];
        bf1[h] = Ws[nLoc + 16][kb + koffB + h];
      }
      acc0 = __builtin_amdgcn_wmma_f32_16x16x32_f16(
          false, afrag, false, bf0, (short)0, acc0, false, false);
      acc1 = __builtin_amdgcn_wmma_f32_16x16x32_f16(
          false, afrag, false, bf1, (short)0, acc1, false, false);
    }
    __syncthreads();
  }

  // C/D layout: lane = N (mod 16), VGPR v = M (+8 for lanes 16-31)
  const int colA = colBase + waveN * 32 + (lane & 15);
  const int colB = colA + 16;
  const int mOff = (lane >= 16) ? 8 : 0;
#pragma unroll
  for (int v = 0; v < 8; ++v) {
    int row = rowBase + waveM * 16 + v + mOff;
    if (row < R) {
      if (colA < O) {
        float y = acc0[v];
        if (act == 1)      y = (y > 0.f) ? y : 0.2f * y;
        else if (act == 2) y = 1.f / (1.f + __expf(-y));
        Y[(long)row * O + colA] = y;
      }
      if (colB < O) {
        float y = acc1[v];
        if (act == 1)      y = (y > 0.f) ? y : 0.2f * y;
        else if (act == 2) y = 1.f / (1.f + __expf(-y));
        Y[(long)row * O + colB] = y;
      }
    }
  }
}

// ============================================================================
// kNN top-k: one wave32 per point. score = dot - 0.5*|m|^2  (ranking of -dist)
// ============================================================================
__global__ void k_topk(const float* __restrict__ dot, const float* __restrict__ x2,
                       int* __restrict__ idx, int N, int Kk)
{
  long wv   = ((long)blockIdx.x * blockDim.x + threadIdx.x) >> 5;
  int  lane = threadIdx.x & 31;
  long b = wv / N, n = wv % N;
  const float* drow = dot + (b * N + n) * (long)N;
  const float* x2b  = x2 + b * N;
  float vals[32];
#pragma unroll
  for (int j = 0; j < 32; ++j) {
    int m = lane + 32 * j;          // N == 1024
    vals[j] = drow[m] - 0.5f * x2b[m];
  }
  for (int kk = 0; kk < Kk; ++kk) {
    float bestV = -3.4e38f; int bestJ = 0;
#pragma unroll
    for (int j = 0; j < 32; ++j)
      if (vals[j] > bestV) { bestV = vals[j]; bestJ = j; }
    int bestM = lane + 32 * bestJ;
    for (int off = 16; off > 0; off >>= 1) {
      float ov = __shfl_xor(bestV, off, 32);
      int   om = __shfl_xor(bestM, off, 32);
      if (ov > bestV || (ov == bestV && om < bestM)) { bestV = ov; bestM = om; }
    }
    if (lane == 0) idx[wv * Kk + kk] = bestM;
    if ((bestM & 31) == lane) vals[bestM >> 5] = -3.4e38f;
  }
}

// ======================= elementwise / gather / pool ========================
__global__ void k_pts(const float* __restrict__ x, float* __restrict__ pts, int B, int N) {
  long t = (long)blockIdx.x * blockDim.x + threadIdx.x;
  long total = (long)B * N * 3; if (t >= total) return;
  int d = t % 3; long bn = t / 3; int n = bn % N; int b = bn / N;
  pts[t] = x[((long)b * 3 + d) * N + n];
}

__global__ void k_x2(const float* __restrict__ f, float* __restrict__ x2, long P, int D) {
  long p = (long)blockIdx.x * blockDim.x + threadIdx.x;
  if (p >= P) return;
  float a = 0.f;
  for (int i = 0; i < D; ++i) { float v = f[p * D + i]; a += v * v; }
  x2[p] = a;
}

__global__ void k_featcat(const float* __restrict__ s, const float* __restrict__ v,
                          float* __restrict__ feat, long P, int Cs, int Cv) {
  int D = Cs + 3 * Cv;
  long t = (long)blockIdx.x * blockDim.x + threadIdx.x;
  long total = P * D; if (t >= total) return;
  int c = t % D; long p = t / D;
  feat[t] = (c < Cs) ? s[p * Cs + c] : v[p * 3 * Cv + (c - Cs)];
}

__global__ void k_init_v(const float* __restrict__ pts, const int* __restrict__ idx,
                         float* __restrict__ v0, int B, int N, int Kk) {
  long p = (long)blockIdx.x * blockDim.x + threadIdx.x;
  long total = (long)B * N * Kk; if (p >= total) return;
  long bn = p / Kk; int b = bn / N;
  int j = idx[p];
  const float* pc = pts + bn * 3;
  const float* pn = pts + ((long)b * N + j) * 3;
  float* o = v0 + p * 6;                 // (p, 3, 2) d-major
#pragma unroll
  for (int d = 0; d < 3; ++d) { o[d * 2 + 0] = pn[d] - pc[d]; o[d * 2 + 1] = pc[d]; }
}

__global__ void k_edge_s(const float* __restrict__ s, const int* __restrict__ idx,
                         float* __restrict__ se, int B, int N, int Kk, int C) {
  long t = (long)blockIdx.x * blockDim.x + threadIdx.x;
  long total = (long)B * N * Kk * 2 * C; if (t >= total) return;
  int c = t % (2 * C); long p = t / (2 * C);
  long bn = p / Kk; int b = bn / N;
  int j = idx[p];
  se[t] = (c < C) ? (s[((long)b * N + j) * C + c] - s[bn * C + c])
                  : s[bn * C + (c - C)];
}

__global__ void k_edge_v(const float* __restrict__ v, const int* __restrict__ idx,
                         float* __restrict__ ve, int B, int N, int Kk, int C) {
  long t = (long)blockIdx.x * blockDim.x + threadIdx.x;
  long total = (long)B * N * Kk * 3 * 2 * C; if (t >= total) return;
  int c = t % (2 * C); long r = t / (2 * C);
  int d = r % 3; long p = r / 3;
  long bn = p / Kk; int b = bn / N;
  int j = idx[p];
  long C3 = 3 * (long)C;
  ve[t] = (c < C) ? (v[((long)b * N + j) * C3 + d * C + c] - v[bn * C3 + d * C + c])
                  : v[bn * C3 + d * C + (c - C)];
}

__global__ void k_norm(const float* __restrict__ v, float* __restrict__ n, long P, int C) {
  long t = (long)blockIdx.x * blockDim.x + threadIdx.x;
  long total = P * C; if (t >= total) return;
  int c = t % C; long p = t / C;
  const float* vp = v + p * 3 * C;
  float a0 = vp[c], a1 = vp[C + c], a2 = vp[2 * C + c];
  n[t] = sqrtf(a0 * a0 + a1 * a1 + a2 * a2 + 1e-8f);
}

__global__ void k_gate(float* __restrict__ v, const float* __restrict__ g, long P, int C) {
  long t = (long)blockIdx.x * blockDim.x + threadIdx.x;
  long total = P * 3 * C; if (t >= total) return;
  int c = t % C; long p = t / (3 * C);
  v[t] *= g[p * C + c];
}

__global__ void k_max_k(const float* __restrict__ in, float* __restrict__ out,
                        long BN, int Kk, int C) {
  long t = (long)blockIdx.x * blockDim.x + threadIdx.x;
  long total = BN * C; if (t >= total) return;
  int c = t % C; long bn = t / C;
  float m = -3.4e38f;
  for (int kk = 0; kk < Kk; ++kk) m = fmaxf(m, in[(bn * Kk + kk) * (long)C + c]);
  out[t] = m;
}

__global__ void k_mean_k(const float* __restrict__ in, float* __restrict__ out,
                         long BN, int Kk, int C) {
  long t = (long)blockIdx.x * blockDim.x + threadIdx.x;
  long total = BN * C; if (t >= total) return;
  int c = t % C; long bn = t / C;
  float a = 0.f;
  for (int kk = 0; kk < Kk; ++kk) a += in[(bn * Kk + kk) * (long)C + c];
  out[t] = a / (float)Kk;
}

__global__ void k_max_n(const float* __restrict__ in, float* __restrict__ out,
                        int B, int N, int C) {
  long t = (long)blockIdx.x * blockDim.x + threadIdx.x;
  long total = (long)B * C; if (t >= total) return;
  int c = t % C; int b = t / C;
  float m = -3.4e38f;
  for (int n = 0; n < N; ++n) m = fmaxf(m, in[((long)b * N + n) * C + c]);
  out[t] = m;
}

__global__ void k_mean_n(const float* __restrict__ in, float* __restrict__ out,
                         int B, int N, int C) {
  long t = (long)blockIdx.x * blockDim.x + threadIdx.x;
  long total = (long)B * C; if (t >= total) return;
  int c = t % C; int b = t / C;
  float a = 0.f;
  for (int n = 0; n < N; ++n) a += in[((long)b * N + n) * C + c];
  out[t] = a / (float)N;
}

__global__ void k_cat4(const float* __restrict__ a, int Ca, const float* __restrict__ b, int Cb,
                       const float* __restrict__ c, int Cc, const float* __restrict__ d, int Cd,
                       float* __restrict__ out, long Rows) {
  int Ct = Ca + Cb + Cc + Cd;
  long t = (long)blockIdx.x * blockDim.x + threadIdx.x;
  long total = Rows * Ct; if (t >= total) return;
  int cc = t % Ct; long r = t / Ct;
  float v;
  if (cc < Ca)                 v = a[r * Ca + cc];
  else if (cc < Ca + Cb)       v = b[r * Cb + cc - Ca];
  else if (cc < Ca + Cb + Cc)  v = c[r * Cc + cc - Ca - Cb];
  else                         v = d[r * Cd + cc - Ca - Cb - Cc];
  out[t] = v;
}

// svfuse: out[p] = [ s[p] (Cs) | pr[c,o]=sum_d v[p,d,c]*z[o,d] flattened (c,o) ]
// z[o,d] = sum_c v[p,d,c]*Wf[o,c]
__global__ void k_svfuse(const float* __restrict__ s, const float* __restrict__ v,
                         const float* __restrict__ Wf, float* __restrict__ out,
                         int Cs, int Cv) {
  long p = blockIdx.x;
  const float* vp = v + p * 3 * (long)Cv;
  __shared__ float z[9];                 // z[o*3+d]
  int tid = threadIdx.x;
  if (tid < 9) {
    int o = tid / 3, d = tid % 3;
    float a = 0.f;
    for (int c = 0; c < Cv; ++c) a += vp[d * Cv + c] * Wf[o * Cv + c];
    z[tid] = a;
  }
  __syncthreads();
  float* op = out + p * (long)(Cs + 3 * Cv);
  for (int i = tid; i < Cs; i += blockDim.x) op[i] = s[p * (long)Cs + i];
  for (int j = tid; j < 3 * Cv; j += blockDim.x) {
    int c = j / 3, o = j % 3;
    float a = 0.f;
#pragma unroll
    for (int d = 0; d < 3; ++d) a += vp[d * Cv + c] * z[o * 3 + d];
    op[Cs + j] = a;
  }
}

// h = [ xg(b) 1016 | xpool(b) 520 | lh(b) 64 | xfine(p) 544 ]  -> (BN, 2144)
__global__ void k_build_h(const float* __restrict__ xg, const float* __restrict__ xpool,
                          const float* __restrict__ lh, const float* __restrict__ xfine,
                          float* __restrict__ h, int N) {
  long t = (long)blockIdx.x * blockDim.x + threadIdx.x;
  long total = (long)4 * N * 2144; if (t >= total) return;
  int c = t % 2144; long p = t / 2144; int b = p / N;
  float v;
  if (c < 1016)      v = xg[(long)b * 1016 + c];
  else if (c < 1536) v = xpool[(long)b * 520 + (c - 1016)];
  else if (c < 1600) v = lh[(long)b * 64 + (c - 1536)];
  else               v = xfine[p * 544 + (c - 1600)];
  h[t] = v;
}

__global__ void k_out_t(const float* __restrict__ y, float* __restrict__ out,
                        int B, int N, int C) {
  long t = (long)blockIdx.x * blockDim.x + threadIdx.x;
  long total = (long)B * C * N; if (t >= total) return;
  int n = t % N; long bc = t / N; int c = bc % C; int b = bc / C;
  out[t] = y[((long)b * N + n) * C + c];
}

// ============================================================================
extern "C" void kernel_launch(void* const* d_in, const int* in_sizes, int n_in,
                              void* d_out, int out_size, void* d_ws, size_t ws_size,
                              hipStream_t stream) {
  (void)in_sizes; (void)n_in; (void)out_size; (void)ws_size;
  const int B_ = 4, N_ = 1024, K_ = 32;
  const long BN = (long)B_ * N_;         // 4096 points
  const long PE = BN * K_;               // 131072 edges

  const float* x    = (const float*)d_in[0];
  const float* l    = (const float*)d_in[1];
  const float* Wv2s = (const float*)d_in[2];
  struct BlkW { const float *Wsv, *Ws, *Wv, *Wg; int Si, Vi, So, Vo; };
  auto P = [&](int i) { return (const float*)d_in[i]; };
  BlkW b1{P(3),  P(4),  P(5),  P(6),    6,   2,  32,  16};
  BlkW b2{P(7),  P(8),  P(9),  P(10),  64,  32,  32,  16};
  BlkW b3{P(11), P(12), P(13), P(14),  64,  32,  64,  24};
  BlkW b4{P(15), P(16), P(17), P(18), 128,  48, 128,  40};
  BlkW b5{P(19), P(20), P(21), P(22), 256,  96, 512, 168};
  BlkW b6{P(23), P(24), P(25), P(26), 512, 168, 256,  88};
  const float* f1 = P(27); const float* f2 = P(28); const float* f3 = P(29);
  const float* W7 = P(30); const float* W8 = P(31); const float* W9 = P(32);
  const float* W10 = P(33); const float* W11 = P(34);

  // ---- bump allocator over d_ws ----
  size_t off = 0;
  auto alloc = [&](long elems) -> float* {
    float* p = (float*)((char*)d_ws + off);
    off += (((size_t)elems * 4) + 255) & ~(size_t)255;
    return p;
  };
  float* pts    = alloc(BN * 3);
  float* x2Buf  = alloc(BN);
  float* dotBuf = alloc((long)B_ * N_ * N_);
  int*   idxBuf = (int*)alloc(PE);
  float* featB  = alloc(BN * 136);
  float* s0     = alloc(PE * 6);
  float* seA    = alloc(PE * 256);   // edge scalar inputs (max 2*128)
  float* veA    = alloc(PE * 288);   // edge vector inputs (max 3*2*48)
  float* vnA    = alloc(PE * 120);   // Wsv / Wv outputs (max 3*40) & b5 reuse
  float* nrmA   = alloc(PE * 40);
  float* gA     = alloc(PE * 40);
  float* seo    = alloc(PE * 128);   // svblock scalar out (max 128)
  float* veo    = alloc(PE * 120);   // svblock vector out (max 3*40)
  float* s1 = alloc(BN * 32);  float* v1 = alloc(BN * 48);
  float* s2 = alloc(BN * 32);  float* v2 = alloc(BN * 48);
  float* s3 = alloc(BN * 64);  float* v3 = alloc(BN * 72);
  float* s4 = alloc(BN * 128); float* v4 = alloc(BN * 120);
  float* scat = alloc(BN * 256); float* vcat = alloc(BN * 288);
  float* xfine = alloc(BN * 544);
  float* s5 = alloc(BN * 512); float* v5 = alloc(BN * 504);
  float* xgf = alloc(BN * 1016); float* xg = alloc(B_ * 1016);
  float* sp  = alloc(B_ * 512);  float* vp = alloc(B_ * 504);
  float* sp2 = alloc(B_ * 256);  float* vp2 = alloc(B_ * 264);
  float* xpool = alloc(B_ * 520);
  float* lh  = alloc(B_ * 64);
  float* h   = alloc(BN * 2144);
  float* h1  = alloc(BN * 256);
  float* h2  = alloc(BN * 256);
  float* h3  = alloc(BN * 128);
  float* yo  = alloc(BN * 50);

  auto ew = [](long total) { return dim3((unsigned)((total + 255) / 256)); };
  auto gemm = [&](const float* X0, int I0, const float* X1, int I1,
                  const float* W, float* Y, long R, int O, int act,
                  int batches = 1, long sX0 = 0, long sX1 = 0, long sW = 0, long sY = 0) {
    dim3 g((unsigned)((O + 63) / 64), (unsigned)((R + 63) / 64), (unsigned)batches);
    k_gemm<<<g, 256, 0, stream>>>(X0, I0, X1, I1, W, Y, (int)R, O, act, sX0, sX1, sW, sY);
  };
  auto knn = [&](const float* feat, int D) {
    k_x2<<<ew(BN), 256, 0, stream>>>(feat, x2Buf, BN, D);
    gemm(feat, D, nullptr, 0, feat, dotBuf, N_, N_, 0,
         B_, (long)N_ * D, 0, (long)N_ * D, (long)N_ * N_);
    k_topk<<<(unsigned)(BN / 8), 256, 0, stream>>>(dotBuf, x2Buf, idxBuf, N_, K_);
  };
  auto svblock = [&](const float* sIn, const float* vIn, long Pn, const BlkW& bw,
                     float* sOut, float* vOut) {
    gemm(vIn, bw.Vi, nullptr, 0, bw.Wsv, vnA, Pn * 3, bw.Vo, 0);
    k_norm<<<ew(Pn * bw.Vo), 256, 0, stream>>>(vnA, nrmA, Pn, bw.Vo);
    gemm(sIn, bw.Si, nrmA, bw.Vo, bw.Ws, sOut, Pn, bw.So, 1);      // lrelu
    gemm(vIn, bw.Vi, nullptr, 0, bw.Wv, vOut, Pn * 3, bw.Vo, 0);
    gemm(sOut, bw.So, nullptr, 0, bw.Wg, gA, Pn, bw.Vo, 2);        // sigmoid
    k_gate<<<ew(Pn * 3 * bw.Vo), 256, 0, stream>>>(vOut, gA, Pn, bw.Vo);
  };

  // ---- stage 0: xyz kNN -> initial equivariant edge features -> b1 ----
  k_pts<<<ew(BN * 3), 256, 0, stream>>>(x, pts, B_, N_);
  knn(pts, 3);
  k_init_v<<<ew(PE), 256, 0, stream>>>(pts, idxBuf, veA, B_, N_, K_);   // v0 (PE,3,2)
  gemm(veA, 2, nullptr, 0, Wv2s, vnA, PE * 3, 6, 0);                    // u (PE,3,6)
  k_norm<<<ew(PE * 6), 256, 0, stream>>>(vnA, s0, PE, 6);               // s0 (PE,6)
  svblock(s0, veA, PE, b1, seo, veo);
  k_max_k<<<ew(BN * 32), 256, 0, stream>>>(seo, s1, BN, K_, 32);
  k_mean_k<<<ew(BN * 48), 256, 0, stream>>>(veo, v1, BN, K_, 48);

  // ---- stages b2..b4: dynamic graph in feature space ----
  struct Layer { const float* s; const float* v; int Cs, Cv; const BlkW* bw; float* so; float* vo; };
  Layer layers[3] = {
    { s1, v1, 32, 16, &b2, s2, v2 },
    { s2, v2, 32, 16, &b3, s3, v3 },
    { s3, v3, 64, 24, &b4, s4, v4 },
  };
  for (int li = 0; li < 3; ++li) {
    const Layer& L = layers[li];
    int D = L.Cs + 3 * L.Cv;
    k_featcat<<<ew(BN * D), 256, 0, stream>>>(L.s, L.v, featB, BN, L.Cs, L.Cv);
    knn(featB, D);
    k_edge_s<<<ew(PE * 2 * L.Cs), 256, 0, stream>>>(L.s, idxBuf, seA, B_, N_, K_, L.Cs);
    k_edge_v<<<ew(PE * 6 * L.Cv), 256, 0, stream>>>(L.v, idxBuf, veA, B_, N_, K_, L.Cv);
    svblock(seA, veA, PE, *L.bw, seo, veo);
    k_max_k<<<ew(BN * L.bw->So), 256, 0, stream>>>(seo, L.so, BN, K_, L.bw->So);
    k_mean_k<<<ew(BN * 3 * L.bw->Vo), 256, 0, stream>>>(veo, L.vo, BN, K_, 3 * L.bw->Vo);
  }

  // ---- multiscale concat, fuse, global head ----
  k_cat4<<<ew(BN * 256), 256, 0, stream>>>(s1, 32, s2, 32, s3, 64, s4, 128, scat, BN);
  k_cat4<<<ew(BN * 3 * 96), 256, 0, stream>>>(v1, 16, v2, 16, v3, 24, v4, 40, vcat, BN * 3);
  k_svfuse<<<(unsigned)BN, 128, 0, stream>>>(scat, vcat, f1, xfine, 256, 96);

  svblock(scat, vcat, BN, b5, s5, v5);
  k_max_n<<<ew((long)B_ * 512), 256, 0, stream>>>(s5, sp, B_, N_, 512);
  k_mean_n<<<ew((long)B_ * 504), 256, 0, stream>>>(v5, vp, B_, N_, 504);
  svblock(sp, vp, B_, b6, sp2, vp2);
  k_svfuse<<<(unsigned)B_, 128, 0, stream>>>(sp2, vp2, f2, xpool, 256, 88);
  k_svfuse<<<(unsigned)BN, 128, 0, stream>>>(s5, v5, f3, xgf, 512, 168);
  k_max_n<<<ew((long)B_ * 1016), 256, 0, stream>>>(xgf, xg, B_, N_, 1016);
  gemm(l, 16, nullptr, 0, W7, lh, B_, 64, 1);                         // lrelu

  k_build_h<<<ew(BN * 2144), 256, 0, stream>>>(xg, xpool, lh, xfine, h, N_);
  gemm(h,  2144, nullptr, 0, W8,  h1, BN, 256, 1);
  gemm(h1, 256,  nullptr, 0, W9,  h2, BN, 256, 1);
  gemm(h2, 256,  nullptr, 0, W10, h3, BN, 128, 1);
  gemm(h3, 128,  nullptr, 0, W11, yo, BN, 50, 0);
  k_out_t<<<ew(BN * 50), 256, 0, stream>>>(yo, (float*)d_out, B_, N_, 50);
}
